// NNBase_42752104464545
// MI455X (gfx1250) — matrix operand order repaired and checked by reading, not compiled, not run on hardware
//
#include <hip/hip_runtime.h>
#include <hip/hip_bf16.h>
#include <math.h>

// Problem constants (match reference)
#define H_   1024
#define A_   8
#define T_   64
#define N_   256
#define H2_  2048
#define H3_  3072
#define KIN_ 2056          // 2H + A
#define KP_  2080          // KIN padded to multiple of 32 (65 * 32)

typedef __attribute__((ext_vector_type(16))) __bf16 v16bf;
typedef __attribute__((ext_vector_type(8)))  float  v8f;

#define WMMA_BF16(C, Amat, Bmat)                                               \
  C = __builtin_amdgcn_wmma_f32_16x16x32_bf16(false, (Amat), false, (Bmat),    \
                                              (short)0, (C), false, false)

__device__ __forceinline__ void split_bf16(float x, __bf16& hi, __bf16& lo) {
  __bf16 h = (__bf16)x;
  hi = h;
  lo = (__bf16)(x - (float)h);
}

// ---------------------------------------------------------------------------
// A-fragment (pre-split bf16, row-major M x Kpad). ISA "16-bit A 16x32":
// lanes 0-15 row M=lane hold K {0..7,16..23}; lanes 16-31 row M=lane-16
// hold K {8..15,24..31}. Two contiguous 16B runs -> 2x b128 loads.
// ---------------------------------------------------------------------------
__device__ __forceinline__ v16bf loadA_frag(const __bf16* __restrict__ A,
                                            int lda, int row0, int k0,
                                            int lane) {
  const int m = row0 + (lane & 15);
  const __bf16* p = A + (size_t)m * lda + k0 + ((lane >> 4) << 3);
  v16bf a;
#pragma unroll
  for (int e = 0; e < 8; ++e) a[e] = p[e];
#pragma unroll
  for (int e = 0; e < 8; ++e) a[8 + e] = p[16 + e];
  return a;
}

// ---------------------------------------------------------------------------
// B-fragment: 32(K) x 16(N) bf16 from split weights (row-major O x Kpad, so
// B[k][o] = W[o][k]). Lanes 0-15 N=lane hold K {0..7,16..23}; lanes
// 16-31 hold K {8..15,24..31}. Two contiguous 16B runs -> 2x b128 loads.
// ---------------------------------------------------------------------------
__device__ __forceinline__ v16bf loadB_frag(const __bf16* __restrict__ B,
                                            int ldb, int col0, int k0,
                                            int lane) {
  const int o = col0 + (lane & 15);
  const __bf16* p = B + (size_t)o * ldb + k0 + ((lane >> 4) << 3);
  v16bf b;
#pragma unroll
  for (int e = 0; e < 16; ++e) b[e] = p[(e & 7) + ((e >> 3) << 4)];
  return b;
}

// ---------------------------------------------------------------------------
// Fused multi-segment GEMM: out = A(256xK) @ W.T + bias, 3xBF16 emulation.
// One wave owns a 16(M) x 64(N) tile: 4 accumulators, 12 WMMA per K-step.
// Tile order is N-major so the 8 waves of a block share identical B loads.
// ---------------------------------------------------------------------------
struct Seg {
  const __bf16* Ahi;
  const __bf16* Alo;
  const __bf16* Bhi;
  const __bf16* Blo;
  const float*  bias;
  float*        out;
  int N, K, lda, ldd, start;   // start = cumulative wave-tile offset
};
struct SegPack { Seg s[5]; int nSeg; int total; };

__global__ __launch_bounds__(256) void gemm_bf16x3(SegPack p) {
  const int wave = blockIdx.x * 8 + (threadIdx.x >> 5);
  if (wave >= p.total) return;           // wave-uniform
  const int lane = threadIdx.x & 31;

  Seg sg = p.s[0];
  if (p.nSeg > 1 && wave >= p.s[1].start) sg = p.s[1];
  if (p.nSeg > 2 && wave >= p.s[2].start) sg = p.s[2];
  if (p.nSeg > 3 && wave >= p.s[3].start) sg = p.s[3];
  if (p.nSeg > 4 && wave >= p.s[4].start) sg = p.s[4];

  const int local = wave - sg.start;
  const int row0  = (local & 15) << 4;   // M tile fastest (M = 256 -> 16 tiles)
  const int col0  = (local >> 4) << 6;   // 64-wide N strip per wave

  v8f c0 = {}, c1 = {}, c2 = {}, c3 = {};

  for (int k0 = 0; k0 < sg.K; k0 += 32) {
    const v16bf ah = loadA_frag(sg.Ahi, sg.lda, row0, k0, lane);
    const v16bf al = loadA_frag(sg.Alo, sg.lda, row0, k0, lane);

    if (k0 + 32 < sg.K) {  // global_prefetch of next K-step B lines
      const size_t nk = (size_t)(col0 + (lane & 15)) * sg.K + k0 + 32;
      __builtin_prefetch(sg.Bhi + nk, 0, 1);
      __builtin_prefetch(sg.Blo + nk, 0, 1);
    }

    {
      const v16bf bh = loadB_frag(sg.Bhi, sg.K, col0, k0, lane);
      const v16bf bl = loadB_frag(sg.Blo, sg.K, col0, k0, lane);
      WMMA_BF16(c0, ah, bh); WMMA_BF16(c0, ah, bl); WMMA_BF16(c0, al, bh);
    }
    {
      const v16bf bh = loadB_frag(sg.Bhi, sg.K, col0 + 16, k0, lane);
      const v16bf bl = loadB_frag(sg.Blo, sg.K, col0 + 16, k0, lane);
      WMMA_BF16(c1, ah, bh); WMMA_BF16(c1, ah, bl); WMMA_BF16(c1, al, bh);
    }
    {
      const v16bf bh = loadB_frag(sg.Bhi, sg.K, col0 + 32, k0, lane);
      const v16bf bl = loadB_frag(sg.Blo, sg.K, col0 + 32, k0, lane);
      WMMA_BF16(c2, ah, bh); WMMA_BF16(c2, ah, bl); WMMA_BF16(c2, al, bh);
    }
    {
      const v16bf bh = loadB_frag(sg.Bhi, sg.K, col0 + 48, k0, lane);
      const v16bf bl = loadB_frag(sg.Blo, sg.K, col0 + 48, k0, lane);
      WMMA_BF16(c3, ah, bh); WMMA_BF16(c3, ah, bl); WMMA_BF16(c3, al, bh);
    }
  }

  // C/D layout: VGPR r -> M = r (lanes 0-15) or 8+r (lanes 16-31); N = lane&15
  const int nA = col0 + (lane & 15);
  const int m0 = row0 + ((lane >> 4) << 3);
  const float b0 = sg.bias[nA];
  const float b1 = sg.bias[nA + 16];
  const float b2 = sg.bias[nA + 32];
  const float b3 = sg.bias[nA + 48];
#pragma unroll
  for (int r = 0; r < 8; ++r) {
    float* o = sg.out + (size_t)(m0 + r) * sg.ldd + nA;
    o[0]  = c0[r] + b0;
    o[16] = c1[r] + b1;
    o[32] = c2[r] + b2;
    o[48] = c3[r] + b3;
  }
}

// ---------------------------------------------------------------------------
// One-time weight split: W (O x K f32) -> hi/lo bf16 (O x Kp, zero padded)
// ---------------------------------------------------------------------------
__global__ void split_w_kernel(const float* __restrict__ W, __bf16* hi,
                               __bf16* lo, int O, int K, int Kp) {
  size_t i = (size_t)blockIdx.x * blockDim.x + threadIdx.x;
  if (i >= (size_t)O * Kp) return;
  int o = (int)(i / Kp), k = (int)(i % Kp);
  float x = (k < K) ? W[(size_t)o * K + k] : 0.0f;
  split_bf16(x, hi[i], lo[i]);
}

// Generic row-block split: X (N_ x cols within ldx) -> hi/lo (N_ x cols)
__global__ void split_rows_kernel(const float* __restrict__ X, int ldx,
                                  int cols, __bf16* hi, __bf16* lo) {
  int i = blockIdx.x * blockDim.x + threadIdx.x;
  if (i >= N_ * cols) return;
  int n = i / cols, k = i % cols;
  split_bf16(X[(size_t)n * ldx + k], hi[i], lo[i]);
}

__global__ void init_state(const float* __restrict__ hxs, float* f, float* h) {
  int i = blockIdx.x * blockDim.x + threadIdx.x;
  if (i >= N_ * H_) return;
  int n = i / H_, j = i % H_;
  f[i] = hxs[(size_t)n * 3 * H_ + j];
  h[i] = hxs[(size_t)n * 3 * H_ + H_ + j];
}

// Build (pre-split to bf16 hi/lo): p_in = [h,q_mu,a]*m, q_in = [f,c_t,a]*m,
// fm = f*m (also f32), hm = h*m (also f32), and split c_t.
__global__ void prep_step(const float* __restrict__ f, const float* __restrict__ h,
                          const float* __restrict__ qmu, int ldq,
                          const float* __restrict__ c_t, const float* __restrict__ a_t,
                          const float* __restrict__ m_t,
                          __bf16* pin_hi, __bf16* pin_lo,
                          __bf16* qin_hi, __bf16* qin_lo,
                          __bf16* fm_hi, __bf16* fm_lo,
                          __bf16* hm_hi, __bf16* hm_lo,
                          __bf16* c_hi, __bf16* c_lo,
                          float* fm, float* hm) {
  int i = blockIdx.x * blockDim.x + threadIdx.x;
  if (i < N_ * KP_) {
    int n = i / KP_, k = i % KP_;
    float m = m_t[n];
    float pv, qv;
    if (k < H_) {
      pv = h[(size_t)n * H_ + k];
      qv = f[(size_t)n * H_ + k];
    } else if (k < 2 * H_) {
      pv = qmu[(size_t)n * ldq + (k - H_)];
      qv = c_t[(size_t)n * H_ + (k - H_)];
    } else if (k < 2 * H_ + A_) {
      float a = a_t[(size_t)n * A_ + (k - 2 * H_)];
      pv = a; qv = a;
    } else {
      pv = 0.0f; qv = 0.0f;     // K padding
    }
    split_bf16(pv * m, pin_hi[i], pin_lo[i]);
    split_bf16(qv * m, qin_hi[i], qin_lo[i]);
  }
  if (i < N_ * H_) {
    float m = m_t[i / H_];
    float fv = f[i] * m;
    float hv = h[i] * m;
    fm[i] = fv;
    hm[i] = hv;
    split_bf16(fv, fm_hi[i], fm_lo[i]);
    split_bf16(hv, hm_hi[i], hm_lo[i]);
    split_bf16(c_t[i], c_hi[i], c_lo[i]);
  }
}

__device__ __forceinline__ float sigf(float x) { return 1.0f / (1.0f + expf(-x)); }

__global__ void gru_combine(const float* __restrict__ f_gi, const float* __restrict__ f_gh,
                            const float* __restrict__ h_gi, const float* __restrict__ h_gh,
                            const float* __restrict__ fm, const float* __restrict__ hm,
                            float* f, float* h) {
  int i = blockIdx.x * blockDim.x + threadIdx.x;
  if (i >= N_ * H_) return;
  int n = i / H_, j = i % H_;
  size_t b = (size_t)n * H3_ + j;
  {  // f-GRU: x = c_t (gi), state = f*m (gh)
    float r  = sigf(f_gi[b] + f_gh[b]);
    float z  = sigf(f_gi[b + H_] + f_gh[b + H_]);
    float nn = tanhf(f_gi[b + 2 * H_] + r * f_gh[b + 2 * H_]);
    f[i] = (1.0f - z) * nn + z * fm[i];
  }
  {  // h-GRU: x = p_mu (gi), state = h*m (gh)
    float r  = sigf(h_gi[b] + h_gh[b]);
    float z  = sigf(h_gi[b + H_] + h_gh[b + H_]);
    float nn = tanhf(h_gi[b + 2 * H_] + r * h_gh[b + 2 * H_]);
    h[i] = (1.0f - z) * nn + z * hm[i];
  }
}

__global__ void final_out(const float* __restrict__ f, const float* __restrict__ h,
                          const float* __restrict__ qlast, float* out) {
  int i = blockIdx.x * blockDim.x + threadIdx.x;
  if (i >= N_ * 3 * H_) return;
  int n = i / (3 * H_), j = i % (3 * H_);
  float v;
  if (j < H_)           v = f[(size_t)n * H_ + j];
  else if (j < 2 * H_)  v = h[(size_t)n * H_ + (j - H_)];
  else                  v = qlast[(size_t)n * H2_ + (j - 2 * H_)];
  out[i] = v;
}

// ---------------------------------------------------------------------------
extern "C" void kernel_launch(void* const* d_in, const int* in_sizes, int n_in,
                              void* d_out, int out_size, void* d_ws, size_t ws_size,
                              hipStream_t stream) {
  (void)in_sizes; (void)n_in; (void)out_size; (void)ws_size;
  const float* c     = (const float*)d_in[0];
  const float* hxs   = (const float*)d_in[1];
  const float* masks = (const float*)d_in[2];
  const float* act   = (const float*)d_in[3];
  const float* hWih  = (const float*)d_in[4];
  const float* hWhh  = (const float*)d_in[5];
  const float* hbih  = (const float*)d_in[6];
  const float* hbhh  = (const float*)d_in[7];
  const float* fWih  = (const float*)d_in[8];
  const float* fWhh  = (const float*)d_in[9];
  const float* fbih  = (const float*)d_in[10];
  const float* fbhh  = (const float*)d_in[11];
  const float* pW    = (const float*)d_in[12];
  const float* pb    = (const float*)d_in[13];
  const float* qW    = (const float*)d_in[14];
  const float* qb    = (const float*)d_in[15];

  float* acc_p   = (float*)d_out;
  float* acc_q   = acc_p + (size_t)T_ * N_ * H2_;
  float* hxs_out = acc_q + (size_t)T_ * N_ * H2_;

  // Workspace carve-up (~105 MB)
  size_t off = 0;
  auto alloc = [&](size_t bytes) -> void* {
    void* p = (char*)d_ws + off;
    off += (bytes + 255) & ~(size_t)255;
    return p;
  };
  __bf16* pW_hi   = (__bf16*)alloc((size_t)H2_ * KP_ * 2);
  __bf16* pW_lo   = (__bf16*)alloc((size_t)H2_ * KP_ * 2);
  __bf16* qW_hi   = (__bf16*)alloc((size_t)H2_ * KP_ * 2);
  __bf16* qW_lo   = (__bf16*)alloc((size_t)H2_ * KP_ * 2);
  __bf16* fWih_hi = (__bf16*)alloc((size_t)H3_ * H_ * 2);
  __bf16* fWih_lo = (__bf16*)alloc((size_t)H3_ * H_ * 2);
  __bf16* fWhh_hi = (__bf16*)alloc((size_t)H3_ * H_ * 2);
  __bf16* fWhh_lo = (__bf16*)alloc((size_t)H3_ * H_ * 2);
  __bf16* hWih_hi = (__bf16*)alloc((size_t)H3_ * H_ * 2);
  __bf16* hWih_lo = (__bf16*)alloc((size_t)H3_ * H_ * 2);
  __bf16* hWhh_hi = (__bf16*)alloc((size_t)H3_ * H_ * 2);
  __bf16* hWhh_lo = (__bf16*)alloc((size_t)H3_ * H_ * 2);
  __bf16* pin_hi  = (__bf16*)alloc((size_t)N_ * KP_ * 2);
  __bf16* pin_lo  = (__bf16*)alloc((size_t)N_ * KP_ * 2);
  __bf16* qin_hi  = (__bf16*)alloc((size_t)N_ * KP_ * 2);
  __bf16* qin_lo  = (__bf16*)alloc((size_t)N_ * KP_ * 2);
  __bf16* fm_hi   = (__bf16*)alloc((size_t)N_ * H_ * 2);
  __bf16* fm_lo   = (__bf16*)alloc((size_t)N_ * H_ * 2);
  __bf16* hm_hi   = (__bf16*)alloc((size_t)N_ * H_ * 2);
  __bf16* hm_lo   = (__bf16*)alloc((size_t)N_ * H_ * 2);
  __bf16* ct_hi   = (__bf16*)alloc((size_t)N_ * H_ * 2);
  __bf16* ct_lo   = (__bf16*)alloc((size_t)N_ * H_ * 2);
  __bf16* pmu_hi  = (__bf16*)alloc((size_t)N_ * H_ * 2);
  __bf16* pmu_lo  = (__bf16*)alloc((size_t)N_ * H_ * 2);
  float* fm   = (float*)alloc((size_t)N_ * H_ * 4);
  float* hm   = (float*)alloc((size_t)N_ * H_ * 4);
  float* fbuf = (float*)alloc((size_t)N_ * H_ * 4);
  float* hbuf = (float*)alloc((size_t)N_ * H_ * 4);
  float* f_gi = (float*)alloc((size_t)N_ * H3_ * 4);
  float* f_gh = (float*)alloc((size_t)N_ * H3_ * 4);
  float* h_gi = (float*)alloc((size_t)N_ * H3_ * 4);
  float* h_gh = (float*)alloc((size_t)N_ * H3_ * 4);

  const int TB = 256;
  auto g1 = [](size_t n) { return (unsigned)((n + 255) / 256); };

  // One-time (per launch) weight hi/lo splits — stay resident in 192MB L2
  split_w_kernel<<<g1((size_t)H2_ * KP_), TB, 0, stream>>>(pW, pW_hi, pW_lo, H2_, KIN_, KP_);
  split_w_kernel<<<g1((size_t)H2_ * KP_), TB, 0, stream>>>(qW, qW_hi, qW_lo, H2_, KIN_, KP_);
  split_w_kernel<<<g1((size_t)H3_ * H_), TB, 0, stream>>>(fWih, fWih_hi, fWih_lo, H3_, H_, H_);
  split_w_kernel<<<g1((size_t)H3_ * H_), TB, 0, stream>>>(fWhh, fWhh_hi, fWhh_lo, H3_, H_, H_);
  split_w_kernel<<<g1((size_t)H3_ * H_), TB, 0, stream>>>(hWih, hWih_hi, hWih_lo, H3_, H_, H_);
  split_w_kernel<<<g1((size_t)H3_ * H_), TB, 0, stream>>>(hWhh, hWhh_hi, hWhh_lo, H3_, H_, H_);

  init_state<<<g1(N_ * H_), TB, 0, stream>>>(hxs, fbuf, hbuf);

  const int tiles2048 = 16 * (H2_ / 64);  // 512 wave-tiles (16x64 per wave)
  const int tiles3072 = 16 * (H3_ / 64);  // 768 wave-tiles

  for (int t = 0; t < T_; ++t) {
    const float* qmu = (t == 0) ? (hxs + 2 * H_)
                                : (acc_q + (size_t)(t - 1) * N_ * H2_);
    const int ldq = (t == 0) ? (3 * H_) : H2_;

    prep_step<<<g1((size_t)N_ * KP_), TB, 0, stream>>>(
        fbuf, hbuf, qmu, ldq, c + (size_t)t * N_ * H_,
        act + (size_t)t * N_ * A_, masks + (size_t)t * N_,
        pin_hi, pin_lo, qin_hi, qin_lo, fm_hi, fm_lo, hm_hi, hm_lo,
        ct_hi, ct_lo, fm, hm);

    // Fused launch: p_dist, q_dist, f_gi, f_gh, h_gh (all independent)
    SegPack P{};
    P.s[0] = { pin_hi, pin_lo, pW_hi, pW_lo, pb,
               acc_p + (size_t)t * N_ * H2_, H2_, KP_, KP_, H2_, 0 };
    P.s[1] = { qin_hi, qin_lo, qW_hi, qW_lo, qb,
               acc_q + (size_t)t * N_ * H2_, H2_, KP_, KP_, H2_, tiles2048 };
    P.s[2] = { ct_hi, ct_lo, fWih_hi, fWih_lo, fbih, f_gi,
               H3_, H_, H_, H3_, 2 * tiles2048 };
    P.s[3] = { fm_hi, fm_lo, fWhh_hi, fWhh_lo, fbhh, f_gh,
               H3_, H_, H_, H3_, 2 * tiles2048 + tiles3072 };
    P.s[4] = { hm_hi, hm_lo, hWhh_hi, hWhh_lo, hbhh, h_gh,
               H3_, H_, H_, H3_, 2 * tiles2048 + 2 * tiles3072 };
    P.nSeg  = 5;
    P.total = 2 * tiles2048 + 3 * tiles3072;           // 3328 -> 416 blocks
    gemm_bf16x3<<<P.total / 8, 256, 0, stream>>>(P);

    // p_mu split, then h_gi = p_mu @ h_Wih.T
    split_rows_kernel<<<g1(N_ * H_), TB, 0, stream>>>(
        acc_p + (size_t)t * N_ * H2_, H2_, H_, pmu_hi, pmu_lo);

    SegPack P2{};
    P2.s[0] = { pmu_hi, pmu_lo, hWih_hi, hWih_lo, hbih, h_gi,
                H3_, H_, H_, H3_, 0 };
    P2.nSeg  = 1;
    P2.total = tiles3072;                              // 96 blocks
    gemm_bf16x3<<<P2.total / 8, 256, 0, stream>>>(P2);

    gru_combine<<<g1(N_ * H_), TB, 0, stream>>>(f_gi, f_gh, h_gi, h_gh,
                                                fm, hm, fbuf, hbuf);
  }

  final_out<<<g1((size_t)N_ * 3 * H_), TB, 0, stream>>>(
      fbuf, hbuf, acc_q + (size_t)(T_ - 1) * N_ * H2_, hxs_out);
}
